// FuseUnit_45148696216017
// MI455X (gfx1250) — compile-verified, matrix operand order checked by reading, not run on hardware
//
#include <hip/hip_runtime.h>

typedef __attribute__((ext_vector_type(16))) __bf16 v16bf;
typedef __attribute__((ext_vector_type(8)))  __bf16 v8bf;
typedef __attribute__((ext_vector_type(8)))  float  v8f;

#define B_ROWS  16384
#define HID     1024
#define OUT_DIM 512

#define BM 128          // rows per block (8 waves x 16)
#define BN 128          // cols per block (8 x 16 sub-tiles per wave)
#define BK 64           // K slice per step (2 x 32-K WMMA sub-steps)
#define LDH 72          // LDS row stride in halves (64 + 8 pad -> conflict-free)
#define KSTEPS (HID / BK)         // 16
#define NGRP_HID (HID / BN)       // 8
#define NGRP_OUT (OUT_DIM / BN)   // 4

#if __has_builtin(__builtin_amdgcn_sched_barrier)
#define SCHED_FENCE() __builtin_amdgcn_sched_barrier(0)
#else
#define SCHED_FENCE() asm volatile("" ::: "memory")
#endif

__device__ inline v8f wmma_bf16(v16bf a, v16bf b, v8f c) {
    return __builtin_amdgcn_wmma_f32_16x16x32_bf16(false, a, false, b, (short)0, c,
                                                   false, false);
}

// ---- load 16 consecutive elements, producing 16 bf16 halves in two v8bf
__device__ inline void load16(const float* __restrict__ p, v8bf& a, v8bf& b) {
    const float4* q = (const float4*)p;
    float4 x0 = q[0], x1 = q[1], x2 = q[2], x3 = q[3];
    a[0] = (__bf16)x0.x; a[1] = (__bf16)x0.y; a[2] = (__bf16)x0.z; a[3] = (__bf16)x0.w;
    a[4] = (__bf16)x1.x; a[5] = (__bf16)x1.y; a[6] = (__bf16)x1.z; a[7] = (__bf16)x1.w;
    b[0] = (__bf16)x2.x; b[1] = (__bf16)x2.y; b[2] = (__bf16)x2.z; b[3] = (__bf16)x2.w;
    b[4] = (__bf16)x3.x; b[5] = (__bf16)x3.y; b[6] = (__bf16)x3.w ? (__bf16)x3.w : (__bf16)x3.w; // placeholder removed below
    b[5] = (__bf16)x3.y; b[6] = (__bf16)x3.z; b[7] = (__bf16)x3.w;
}
__device__ inline void load16(const __bf16* __restrict__ p, v8bf& a, v8bf& b) {
    const v8bf* q = (const v8bf*)p;
    a = q[0];
    b = q[1];
}

// ---- WMMA fragment from a 32-wide window of an LDS slice row (CDNA5 16-bit
//      A/B layout: lane group g = lane/16 holds K in [8g,8g+8) and [16+8g,+8))
__device__ inline v16bf lds_frag(const __bf16* __restrict__ rowp, int grp) {
    v8bf lo = *(const v8bf*)(rowp + 8 * grp);
    v8bf hi = *(const v8bf*)(rowp + 16 + 8 * grp);
    v16bf r;
#pragma unroll
    for (int e = 0; e < 8; ++e) {
        r[e]     = lo[e];
        r[8 + e] = hi[e];
    }
    return r;
}

// ---- block-cooperative, double-buffered GEMM over K=HID for one 128x128 tile.
// Each wave accumulates its 16(M) x 128(N) strip into acc[8].
template <typename TA>
__device__ inline void gemm_tile(const TA* __restrict__ A, int rowBlk,
                                 const __bf16* __restrict__ W, int nBase,
                                 __bf16* __restrict__ lsA, __bf16* __restrict__ lsB,
                                 v8f acc[8], int tid)
{
    const int lane = tid & 31;
    const int wave = tid >> 5;
    const int grp  = lane >> 4;
    const int lr   = lane & 15;
    const int ldRow  = tid >> 1;   // 0..127: row this thread stages
    const int ldHalf = tid & 1;    // which 32-wide half of the 64-wide slice

    const TA*     aPtr = A + (size_t)(rowBlk + ldRow) * HID + 32 * ldHalf;
    const __bf16* bPtr = W + (size_t)(nBase  + ldRow) * HID + 32 * ldHalf;

    v8bf a[4], b[4];
    load16(aPtr,      a[0], a[1]);
    load16(aPtr + 16, a[2], a[3]);
    load16(bPtr,      b[0], b[1]);
    load16(bPtr + 16, b[2], b[3]);

    const int stO     = ldRow * LDH + 32 * ldHalf;  // LDS store offset (halves)
    const int aRowOff = (wave * 16 + lr) * LDH;

    for (int ks = 0; ks < KSTEPS; ++ks) {
        __bf16* bufA = lsA + (ks & 1) * (BM * LDH);
        __bf16* bufB = lsB + (ks & 1) * (BN * LDH);
#pragma unroll
        for (int q = 0; q < 4; ++q) {
            *(v8bf*)(bufA + stO + 8 * q) = a[q];
            *(v8bf*)(bufB + stO + 8 * q) = b[q];
        }
        __syncthreads();
        if (ks + 1 < KSTEPS) {  // prefetch next slice into registers
            load16(aPtr + (ks + 1) * BK,      a[0], a[1]);
            load16(aPtr + (ks + 1) * BK + 16, a[2], a[3]);
            load16(bPtr + (ks + 1) * BK,      b[0], b[1]);
            load16(bPtr + (ks + 1) * BK + 16, b[2], b[3]);
        }
        SCHED_FENCE();  // pin the prefetch issue point above the math

        v16bf af0 = lds_frag(bufA + aRowOff, grp);
        v16bf af1 = lds_frag(bufA + aRowOff + 32, grp);
        // one-fragment lookahead on B so each WMMA only waits on its own pair
        v16bf bc0 = lds_frag(bufB + lr * LDH, grp);
        v16bf bc1 = lds_frag(bufB + lr * LDH + 32, grp);
#pragma unroll
        for (int j = 0; j < 8; ++j) {
            v16bf bn0 = bc0, bn1 = bc1;
            if (j < 7) {
                bn0 = lds_frag(bufB + ((j + 1) * 16 + lr) * LDH, grp);
                bn1 = lds_frag(bufB + ((j + 1) * 16 + lr) * LDH + 32, grp);
            }
            acc[j] = wmma_bf16(af0, bc0, acc[j]);
            acc[j] = wmma_bf16(af1, bc1, acc[j]);
            bc0 = bn0;
            bc1 = bn1;
        }
        __syncthreads();
    }
}

// ---- fp32 -> bf16 pre-conversion
__global__ void f32_to_bf16(const float* __restrict__ in, __bf16* __restrict__ out, int n) {
    int i = blockIdx.x * blockDim.x + threadIdx.x;
    if (i < n) out[i] = (__bf16)in[i];
}

// ---- Phase A: trans = s@W1^T+b1 fragments -> per-(row, N-group) partial dot.
// Each (rowBlk, gIdx) block writes dotPart[gIdx][row] exactly once (no atomics).
template <typename TA>
__global__ __launch_bounds__(256) void phaseA_dot(
    const TA* __restrict__ s, const float* __restrict__ sF,
    const __bf16* __restrict__ w1B, const float* __restrict__ b1,
    float* __restrict__ dotPart)
{
    __shared__ __bf16 lsA[2 * BM * LDH];
    __shared__ __bf16 lsB[2 * BN * LDH];

    const int tid  = threadIdx.x;
    const int lane = tid & 31;
    const int wave = tid >> 5;
    const int grp  = lane >> 4;
    const int lr   = lane & 15;
    const int rowBlk  = blockIdx.x * BM;
    const int rowWave = rowBlk + wave * 16;
    const int gIdx    = blockIdx.y;

    v8f acc[8];
#pragma unroll
    for (int j = 0; j < 8; ++j)
#pragma unroll
        for (int v = 0; v < 8; ++v) acc[j][v] = 0.f;

    gemm_tile<TA>(s, rowBlk, w1B, gIdx * BN, lsA, lsB, acc, tid);

    float dacc[8];
#pragma unroll
    for (int v = 0; v < 8; ++v) dacc[v] = 0.f;
#pragma unroll
    for (int j = 0; j < 8; ++j) {
        const int   n    = gIdx * BN + j * 16 + lr;
        const float bias = b1[n];
#pragma unroll
        for (int v = 0; v < 8; ++v) {
            const int m = rowWave + v + 8 * grp;
            dacc[v] += (acc[j][v] + bias) * sF[(size_t)m * HID + n];
        }
    }
    // width-16 reduction: lane-group g ends up with partial dot for rows v+8g
#pragma unroll
    for (int off = 1; off < 16; off <<= 1)
#pragma unroll
        for (int v = 0; v < 8; ++v)
            dacc[v] += __shfl_xor(dacc[v], off, 16);

    if (lr == 0) {
#pragma unroll
        for (int v = 0; v < 8; ++v)
            dotPart[(size_t)gIdx * B_ROWS + rowWave + 8 * grp + v] = dacc[v];
    }
}

// ---- Phase B: combine = combin@W2^T+b2, epilogue relu(x*scale*dot) -> bf16 act
template <typename TA>
__global__ __launch_bounds__(256) void phaseB_act(
    const TA* __restrict__ combin,
    const __bf16* __restrict__ w2B, const float* __restrict__ b2,
    const float* __restrict__ dotPart, __bf16* __restrict__ act)
{
    __shared__ __bf16 lsA[2 * BM * LDH];
    __shared__ __bf16 lsB[2 * BN * LDH];

    const int tid  = threadIdx.x;
    const int lane = tid & 31;
    const int wave = tid >> 5;
    const int grp  = lane >> 4;
    const int lr   = lane & 15;
    const int rowBlk  = blockIdx.x * BM;
    const int rowWave = rowBlk + wave * 16;
    const int gIdx    = blockIdx.y;

    // finalize dot for this lane's 8 rows (sum of 8 deterministic partials)
    float sc[8];
#pragma unroll
    for (int v = 0; v < 8; ++v) {
        const int m = rowWave + v + 8 * grp;
        float d = 0.f;
#pragma unroll
        for (int g2 = 0; g2 < NGRP_HID; ++g2)
            d += dotPart[(size_t)g2 * B_ROWS + m];
        sc[v] = d * 0.03125f;  // 1/sqrt(1024)
    }

    v8f acc[8];
#pragma unroll
    for (int j = 0; j < 8; ++j)
#pragma unroll
        for (int v = 0; v < 8; ++v) acc[j][v] = 0.f;

    gemm_tile<TA>(combin, rowBlk, w2B, gIdx * BN, lsA, lsB, acc, tid);

#pragma unroll
    for (int j = 0; j < 8; ++j) {
        const int   n    = gIdx * BN + j * 16 + lr;
        const float bias = b2[n];
#pragma unroll
        for (int v = 0; v < 8; ++v) {
            const int   m   = rowWave + v + 8 * grp;
            const float val = fmaxf((acc[j][v] + bias) * sc[v], 0.f);
            act[(size_t)m * HID + n] = (__bf16)val;
        }
    }
}

// ---- Stage 2: out = act @ W3^T + b3
__global__ __launch_bounds__(256) void stage2_out(
    const __bf16* __restrict__ act, const __bf16* __restrict__ w3B,
    const float* __restrict__ b3, float* __restrict__ out)
{
    __shared__ __bf16 lsA[2 * BM * LDH];
    __shared__ __bf16 lsB[2 * BN * LDH];

    const int tid  = threadIdx.x;
    const int lane = tid & 31;
    const int wave = tid >> 5;
    const int grp  = lane >> 4;
    const int lr   = lane & 15;
    const int rowBlk  = blockIdx.x * BM;
    const int rowWave = rowBlk + wave * 16;
    const int gIdx    = blockIdx.y;

    v8f acc[8];
#pragma unroll
    for (int j = 0; j < 8; ++j)
#pragma unroll
        for (int v = 0; v < 8; ++v) acc[j][v] = 0.f;

    gemm_tile<__bf16>(act, rowBlk, w3B, gIdx * BN, lsA, lsB, acc, tid);

#pragma unroll
    for (int j = 0; j < 8; ++j) {
        const int   n    = gIdx * BN + j * 16 + lr;
        const float bias = b3[n];
#pragma unroll
        for (int v = 0; v < 8; ++v) {
            const int m = rowWave + v + 8 * grp;
            out[(size_t)m * OUT_DIM + n] = acc[j][v] + bias;
        }
    }
}

extern "C" void kernel_launch(void* const* d_in, const int* in_sizes, int n_in,
                              void* d_out, int out_size, void* d_ws, size_t ws_size,
                              hipStream_t stream) {
    const float* combin = (const float*)d_in[0];
    const float* s      = (const float*)d_in[1];
    const float* W1     = (const float*)d_in[2];
    const float* b1     = (const float*)d_in[3];
    const float* W2     = (const float*)d_in[4];
    const float* b2     = (const float*)d_in[5];
    const float* W3     = (const float*)d_in[6];
    const float* b3     = (const float*)d_in[7];
    float* out = (float*)d_out;

    char* ws = (char*)d_ws;
    const size_t MB = 1024ull * 1024ull;
    __bf16* w1B     = (__bf16*)(ws + 0 * MB);    // 2 MB
    __bf16* w2B     = (__bf16*)(ws + 2 * MB);    // 2 MB
    __bf16* w3B     = (__bf16*)(ws + 4 * MB);    // 1 MB
    __bf16* act     = (__bf16*)(ws + 5 * MB);    // 32 MB
    float*  dotPart = (float*)(ws + 37 * MB);    // 8 * 16384 * 4 = 512 KB
    __bf16* sB      = (__bf16*)(ws + 38 * MB);   // 32 MB (optional)
    __bf16* cB      = (__bf16*)(ws + 70 * MB);   // 32 MB (optional)

    const int nW12 = HID * HID;
    const int nW3  = OUT_DIM * HID;
    f32_to_bf16<<<(nW12 + 255) / 256, 256, 0, stream>>>(W1, w1B, nW12);
    f32_to_bf16<<<(nW12 + 255) / 256, 256, 0, stream>>>(W2, w2B, nW12);
    f32_to_bf16<<<(nW3  + 255) / 256, 256, 0, stream>>>(W3, w3B, nW3);

    const dim3 gridS1(B_ROWS / BM, NGRP_HID);   // 128 x 8
    const dim3 gridS2(B_ROWS / BM, NGRP_OUT);   // 128 x 4

    const bool preconv = (ws_size >= 102 * MB);
    if (preconv) {
        const int nIn = B_ROWS * HID;
        f32_to_bf16<<<(nIn + 255) / 256, 256, 0, stream>>>(s,      sB, nIn);
        f32_to_bf16<<<(nIn + 255) / 256, 256, 0, stream>>>(combin, cB, nIn);
        phaseA_dot<__bf16><<<gridS1, 256, 0, stream>>>(sB, s, w1B, b1, dotPart);
        phaseB_act<__bf16><<<gridS1, 256, 0, stream>>>(cB, w2B, b2, dotPart, act);
    } else {
        phaseA_dot<float><<<gridS1, 256, 0, stream>>>(s, s, w1B, b1, dotPart);
        phaseB_act<float><<<gridS1, 256, 0, stream>>>(combin, w2B, b2, dotPart, act);
    }
    stage2_out<<<gridS2, 256, 0, stream>>>(act, w3B, b3, out);
}
